// PrivateSmurf_1065151889632
// MI455X (gfx1250) — matrix operand order repaired and checked by reading, not compiled, not run on hardware
//
#include <hip/hip_runtime.h>
#include <hip/hip_bf16.h>
#include <math.h>

#define NN 20000          // nodes
#define NE 320000         // edges
#define NIN 128           // node input dim
#define HC  256           // heads*hidden
#define TD  64            // time dim
#define NHEADS 4

typedef __attribute__((ext_vector_type(16))) __bf16 v16bf;
typedef __attribute__((ext_vector_type(8)))  __bf16 v8bf;
typedef __attribute__((ext_vector_type(8)))  float  v8f;

// ---------- converts ----------

// W [K,N] f32 row-major  ->  Wt [N,K] bf16 (B-operand, column-contiguous per output channel)
__global__ __launch_bounds__(256) void k_wt_bf16(const float* __restrict__ W,
                                                 __bf16* __restrict__ Wt, int K, int N) {
    int idx = blockIdx.x * blockDim.x + threadIdx.x;
    if (idx >= K * N) return;
    int k = idx / N, n = idx - k * N;
    Wt[(size_t)n * K + k] = (__bf16)W[idx];
}

__global__ __launch_bounds__(256) void k_f32_to_bf16(const float* __restrict__ in,
                                                     __bf16* __restrict__ out, int n) {
    int idx = blockIdx.x * blockDim.x + threadIdx.x;
    if (idx < n) out[idx] = (__bf16)in[idx];
}

// time encoding: t = raw+noise; te[:,2d]=sin(t*w_d), te[:,2d+1]=cos(t*w_d); straight to bf16
__global__ __launch_bounds__(256) void k_time_encode(const float* __restrict__ rt,
                                                     const float* __restrict__ noise,
                                                     __bf16* __restrict__ te) {
    int idx = blockIdx.x * blockDim.x + threadIdx.x;
    if (idx >= NE * 32) return;
    int e = idx >> 5, d = idx & 31;
    float t = rt[e] + noise[e];
    float div = __expf((float)(2 * d) * (-9.210340371976184f / (float)TD)); // -ln(1e4)/TD
    float arg = t * div;
    te[(size_t)e * TD + 2 * d]     = (__bf16)__sinf(arg);
    te[(size_t)e * TD + 2 * d + 1] = (__bf16)__cosf(arg);
}

// ---------- WMMA bf16 GEMM:  C[M,N] = A[M,K](bf16) * B(bf16, stored [N,K]) + bias ----------
// 1 wave -> one 16x16 tile; 8 waves/block -> 128 output columns; grid(M/16, N/128)
__global__ __launch_bounds__(256) void k_gemm_bf16(const __bf16* __restrict__ A,
                                                   const __bf16* __restrict__ Bt,
                                                   const float* __restrict__ bias,
                                                   float* __restrict__ C,
                                                   int M, int N, int K) {
    int wave = threadIdx.x >> 5;
    int lane = threadIdx.x & 31;
    int m0 = blockIdx.x * 16;
    int n0 = (blockIdx.y * 8 + wave) * 16;
    if (n0 >= N || m0 >= M) return;            // wave-uniform

    int ln15 = lane & 15;
    int hi   = lane >> 4;

    // A fragment (16x32 bf16): lane = row ln15; VGPR0-3 hold K = hi*8+0..7 (pairs),
    // VGPR4-7 hold K = 16+hi*8+0..7  -> two contiguous 16-byte runs per lane.
    const __bf16* arow = A + (size_t)(m0 + ln15) * K + hi * 8;
    // B fragment (32x16 bf16): lane = col ln15; per lane K = hi*8+0..7 and 16+hi*8+0..7 (same as A)
    // because B is stored transposed [N][K].
    const __bf16* bcol = Bt + (size_t)(n0 + ln15) * K + hi * 8;

    v8f acc = {};
    for (int k0 = 0; k0 < K; k0 += 32) {
        v8bf alo = *(const v8bf*)(arow + k0);
        v8bf ahi = *(const v8bf*)(arow + k0 + 16);
        v8bf b0  = *(const v8bf*)(bcol + k0);
        v8bf b1  = *(const v8bf*)(bcol + k0 + 16);
        v16bf a, b;
#pragma unroll
        for (int i = 0; i < 8; i++) { a[i] = alo[i]; a[8 + i] = ahi[i]; }
#pragma unroll
        for (int i = 0; i < 8; i++) { b[i] = b0[i];  b[8 + i] = b1[i];  }
        acc = __builtin_amdgcn_wmma_f32_16x16x32_bf16(
            /*neg_a=*/false, a, /*neg_b=*/false, b,
            /*c_mod=*/(short)0, acc, /*reuse_a=*/false, /*reuse_b=*/false);
    }

    // C/D layout: lanes 0-15 N=lane, rows 0..7 in elems 0..7; lanes 16-31 rows 8..15
    float bs = bias ? bias[n0 + ln15] : 0.0f;
#pragma unroll
    for (int i = 0; i < 8; i++) {
        int m = m0 + hi * 8 + i;
        C[(size_t)m * N + n0 + ln15] = acc[i] + bs;
    }
}

// ---------- edge attention ----------

// wave per edge: alpha[e,h] = (q[dst] . (k[src]+tproj[e]))_h / 8 ; atomic segment-max
__global__ __launch_bounds__(256) void k_edge_alpha(const long long* __restrict__ src,
                                                    const long long* __restrict__ dst,
                                                    const float* __restrict__ q,
                                                    const float* __restrict__ k,
                                                    const float* __restrict__ tp,
                                                    float* __restrict__ alpha,
                                                    unsigned int* __restrict__ nmax) {
    int e = blockIdx.x * (blockDim.x >> 5) + (threadIdx.x >> 5);
    if (e >= NE) return;
    int lane = threadIdx.x & 31;
    long long s = src[e], d = dst[e];
    const float* qr = q  + (size_t)d * HC;
    const float* kr = k  + (size_t)s * HC;
    const float* tr = tp + (size_t)e * HC;
    int base = lane * 8;
    float p = 0.0f;
#pragma unroll
    for (int i = 0; i < 8; i++) {
        int c = base + i;
        p += qr[c] * (kr[c] + tr[c]);
    }
    // reduce within 8-lane head groups
    p += __shfl_xor(p, 1, 32);
    p += __shfl_xor(p, 2, 32);
    p += __shfl_xor(p, 4, 32);
    if ((lane & 7) == 0) {
        int h = lane >> 3;
        float a = p * 0.125f;                           // 1/sqrt(64)
        alpha[(size_t)e * NHEADS + h] = a;
        unsigned int bits = __float_as_uint(a);
        unsigned int key  = (bits & 0x80000000u) ? ~bits : (bits | 0x80000000u);
        atomicMax(nmax + (size_t)d * NHEADS + h, key);
    }
}

// thread per (edge,head): ea = exp(alpha - max[dst]); atomic segment-sum
__global__ __launch_bounds__(256) void k_edge_expsum(const long long* __restrict__ dst,
                                                     float* __restrict__ alpha,
                                                     const unsigned int* __restrict__ nmax,
                                                     float* __restrict__ denom) {
    int idx = blockIdx.x * blockDim.x + threadIdx.x;
    if (idx >= NE * NHEADS) return;
    int e = idx >> 2, h = idx & 3;
    long long d = dst[e];
    unsigned int key  = nmax[(size_t)d * NHEADS + h];
    unsigned int bits = (key & 0x80000000u) ? (key & 0x7FFFFFFFu) : ~key;
    float amax = __uint_as_float(bits);
    float ea = __expf(alpha[idx] - amax);
    alpha[idx] = ea;
    atomicAdd(denom + (size_t)d * NHEADS + h, ea);
}

// thread per (edge, 4 channels): agg[dst] += v[src] * ea/denom[dst]
__global__ __launch_bounds__(256) void k_edge_scatter(const long long* __restrict__ src,
                                                      const long long* __restrict__ dst,
                                                      const float* __restrict__ v,
                                                      const float* __restrict__ alpha,
                                                      const float* __restrict__ denom,
                                                      float* __restrict__ agg) {
    int idx = blockIdx.x * blockDim.x + threadIdx.x;
    if (idx >= NE * (HC / 4)) return;
    int e  = idx >> 6;
    int c4 = (idx & 63) * 4;
    int h  = c4 >> 6;
    long long s = src[e], d = dst[e];
    float w = alpha[(size_t)e * NHEADS + h] / (denom[(size_t)d * NHEADS + h] + 1e-16f);
    float4 vv = *(const float4*)(v + (size_t)s * HC + c4);
    float* out = agg + (size_t)d * HC + c4;
    atomicAdd(out + 0, vv.x * w);
    atomicAdd(out + 1, vv.y * w);
    atomicAdd(out + 2, vv.z * w);
    atomicAdd(out + 3, vv.w * w);
}

__global__ __launch_bounds__(256) void k_relu(const float* __restrict__ agg,
                                              float* __restrict__ xf,
                                              __bf16* __restrict__ xb, int n) {
    int idx = blockIdx.x * blockDim.x + threadIdx.x;
    if (idx >= n) return;
    float r = fmaxf(agg[idx], 0.0f);
    if (xf) xf[idx] = r;
    if (xb) xb[idx] = (__bf16)r;
}

// wave per edge: out[e] = (x[src]+x[dst]) . wc + bc
__global__ __launch_bounds__(256) void k_edge_out(const long long* __restrict__ src,
                                                  const long long* __restrict__ dst,
                                                  const float* __restrict__ x,
                                                  const float* __restrict__ wc,
                                                  const float* __restrict__ bc,
                                                  float* __restrict__ out) {
    int e = blockIdx.x * (blockDim.x >> 5) + (threadIdx.x >> 5);
    if (e >= NE) return;
    int lane = threadIdx.x & 31;
    long long s = src[e], d = dst[e];
    const float* xs = x + (size_t)s * HC;
    const float* xd = x + (size_t)d * HC;
    float p = 0.0f;
#pragma unroll
    for (int i = 0; i < 8; i++) {
        int c = lane * 8 + i;
        p += (xs[c] + xd[c]) * wc[c];
    }
    for (int m = 1; m < 32; m <<= 1) p += __shfl_xor(p, m, 32);
    if (lane == 0) out[e] = p + bc[0];
}

// ---------- host ----------

extern "C" void kernel_launch(void* const* d_in, const int* in_sizes, int n_in,
                              void* d_out, int out_size, void* d_ws, size_t ws_size,
                              hipStream_t stream) {
    const long long* edge_index = (const long long*)d_in[0];
    const long long* src = edge_index;
    const long long* dst = edge_index + NE;
    const float* raw_time = (const float*)d_in[1];
    const float* noise    = (const float*)d_in[2];
    const float* node_emb = (const float*)d_in[3];
    const float* wq1 = (const float*)d_in[4];  const float* bq1 = (const float*)d_in[5];
    const float* wk1 = (const float*)d_in[6];  const float* bk1 = (const float*)d_in[7];
    const float* wv1 = (const float*)d_in[8];  const float* bv1 = (const float*)d_in[9];
    const float* wt1 = (const float*)d_in[10]; const float* bt1 = (const float*)d_in[11];
    const float* wq2 = (const float*)d_in[12]; const float* bq2 = (const float*)d_in[13];
    const float* wk2 = (const float*)d_in[14]; const float* bk2 = (const float*)d_in[15];
    const float* wv2 = (const float*)d_in[16]; const float* bv2 = (const float*)d_in[17];
    const float* wt2 = (const float*)d_in[18]; const float* bt2 = (const float*)d_in[19];
    const float* wc  = (const float*)d_in[20]; const float* bc  = (const float*)d_in[21];
    float* out = (float*)d_out;

    // -------- workspace layout (256B aligned) --------
    char* ws = (char*)d_ws;
    size_t off = 0;
    auto alloc = [&](size_t bytes) -> char* {
        char* p = ws + off;
        off += (bytes + 255) & ~(size_t)255;
        return p;
    };
    __bf16* te_b   = (__bf16*)alloc((size_t)NE * TD * 2);     // bf16 time encoding [E,64]
    float*  tproj  = (float*)alloc((size_t)NE * HC * 4);      // projected time [E,256] (reused per layer)
    __bf16* emb_b  = (__bf16*)alloc((size_t)NN * NIN * 2);
    __bf16* x1b    = (__bf16*)alloc((size_t)NN * HC * 2);
    float*  x2f    = (float*)alloc((size_t)NN * HC * 4);
    float*  qf     = (float*)alloc((size_t)NN * HC * 4);
    float*  kf     = (float*)alloc((size_t)NN * HC * 4);
    float*  vf     = (float*)alloc((size_t)NN * HC * 4);
    float*  agg    = (float*)alloc((size_t)NN * HC * 4);
    float*  alpha  = (float*)alloc((size_t)NE * NHEADS * 4);
    unsigned int* nmax = (unsigned int*)alloc((size_t)NN * NHEADS * 4);
    float*  denom  = (float*)alloc((size_t)NN * NHEADS * 4);
    __bf16* wq1t = (__bf16*)alloc((size_t)NIN * HC * 2);
    __bf16* wk1t = (__bf16*)alloc((size_t)NIN * HC * 2);
    __bf16* wv1t = (__bf16*)alloc((size_t)NIN * HC * 2);
    __bf16* wt1t = (__bf16*)alloc((size_t)TD * HC * 2);
    __bf16* wq2t = (__bf16*)alloc((size_t)HC * HC * 2);
    __bf16* wk2t = (__bf16*)alloc((size_t)HC * HC * 2);
    __bf16* wv2t = (__bf16*)alloc((size_t)HC * HC * 2);
    __bf16* wt2t = (__bf16*)alloc((size_t)TD * HC * 2);
    (void)ws_size; (void)n_in; (void)in_sizes; (void)out_size;

    // -------- weight / activation converts --------
    auto cvtW = [&](const float* W, __bf16* Wt, int K, int N) {
        k_wt_bf16<<<(K * N + 255) / 256, 256, 0, stream>>>(W, Wt, K, N);
    };
    cvtW(wq1, wq1t, NIN, HC); cvtW(wk1, wk1t, NIN, HC); cvtW(wv1, wv1t, NIN, HC);
    cvtW(wt1, wt1t, TD, HC);
    cvtW(wq2, wq2t, HC, HC);  cvtW(wk2, wk2t, HC, HC);  cvtW(wv2, wv2t, HC, HC);
    cvtW(wt2, wt2t, TD, HC);
    k_f32_to_bf16<<<(NN * NIN + 255) / 256, 256, 0, stream>>>(node_emb, emb_b, NN * NIN);
    k_time_encode<<<(NE * 32 + 255) / 256, 256, 0, stream>>>(raw_time, noise, te_b);

    auto gemm = [&](const __bf16* A, const __bf16* Bt, const float* bias, float* C,
                    int M, int N, int K) {
        dim3 grid(M / 16, N / 128);
        k_gemm_bf16<<<grid, 256, 0, stream>>>(A, Bt, bias, C, M, N, K);
    };

    auto attn_layer = [&](const __bf16* xin_b, int Kin,
                          const __bf16* wq_t, const float* bq,
                          const __bf16* wk_t, const float* bk,
                          const __bf16* wv_t, const float* bv,
                          const __bf16* wt_t, const float* bt,
                          float* x_out_f, __bf16* x_out_b) {
        gemm(xin_b, wq_t, bq, qf, NN, HC, Kin);
        gemm(xin_b, wk_t, bk, kf, NN, HC, Kin);
        gemm(xin_b, wv_t, bv, vf, NN, HC, Kin);
        gemm(te_b,  wt_t, bt, tproj, NE, HC, TD);
        hipMemsetAsync(nmax,  0, (size_t)NN * NHEADS * 4, stream);
        hipMemsetAsync(denom, 0, (size_t)NN * NHEADS * 4, stream);
        hipMemsetAsync(agg,   0, (size_t)NN * HC * 4, stream);
        k_edge_alpha<<<(NE + 7) / 8, 256, 0, stream>>>(src, dst, qf, kf, tproj, alpha, nmax);
        k_edge_expsum<<<(NE * NHEADS + 255) / 256, 256, 0, stream>>>(dst, alpha, nmax, denom);
        k_edge_scatter<<<(NE * (HC / 4) + 255) / 256, 256, 0, stream>>>(src, dst, vf, alpha, denom, agg);
        k_relu<<<(NN * HC + 255) / 256, 256, 0, stream>>>(agg, x_out_f, x_out_b, NN * HC);
    };

    // layer 1: input node_emb(bf16) [N,128] -> x1 (bf16 only; f32 not needed downstream)
    attn_layer(emb_b, NIN, wq1t, bq1, wk1t, bk1, wv1t, bv1, wt1t, bt1,
               (float*)nullptr, x1b);
    // layer 2: input x1(bf16) [N,256] -> x2 (f32 only)
    attn_layer(x1b, HC, wq2t, bq2, wk2t, bk2, wv2t, bv2, wt2t, bt2,
               x2f, (__bf16*)nullptr);

    // final edge readout
    k_edge_out<<<(NE + 7) / 8, 256, 0, stream>>>(src, dst, x2f, wc, bc, out);
}
